// Attention_51496657879647
// MI455X (gfx1250) — matrix operand order repaired and checked by reading, not compiled, not run on hardware
//
#include <hip/hip_runtime.h>

#define NCTX   2048
#define DMODEL 1024
#define NHEAD  16
#define DHEAD  64
#define ROTD   32
#define QK_SCALE 0.125f            // 64^-0.5
#define NEG_INF (-3.402823466e+38f)

typedef __attribute__((ext_vector_type(16))) _Float16 v16h;
typedef __attribute__((ext_vector_type(8)))  float    v8f;

union AF { uint4 q[2]; v16h v; };

__device__ __forceinline__ v16h pack16(float4 a, float4 b, float4 c, float4 d) {
  v16h v;
  v[0]=(_Float16)a.x;  v[1]=(_Float16)a.y;  v[2]=(_Float16)a.z;  v[3]=(_Float16)a.w;
  v[4]=(_Float16)b.x;  v[5]=(_Float16)b.y;  v[6]=(_Float16)b.z;  v[7]=(_Float16)b.w;
  v[8]=(_Float16)c.x;  v[9]=(_Float16)c.y;  v[10]=(_Float16)c.z; v[11]=(_Float16)c.w;
  v[12]=(_Float16)d.x; v[13]=(_Float16)d.y; v[14]=(_Float16)d.z; v[15]=(_Float16)d.w;
  return v;
}

// ---------------- LayerNorm -> f16 ----------------
__global__ void __launch_bounds__(256) ln_kernel(const float* __restrict__ x,
                                                 const float* __restrict__ g,
                                                 const float* __restrict__ b,
                                                 _Float16* __restrict__ xnh) {
  __shared__ float red[256];
  const int n = blockIdx.x, t = threadIdx.x;
  const float* p = x + (size_t)n * DMODEL;
  float v[4]; float s = 0.f;
  #pragma unroll
  for (int i = 0; i < 4; ++i) { v[i] = p[t + i*256]; s += v[i]; }
  red[t] = s; __syncthreads();
  for (int st = 128; st > 0; st >>= 1) { if (t < st) red[t] += red[t+st]; __syncthreads(); }
  const float mu = red[0] * (1.0f / DMODEL);
  __syncthreads();
  float sq = 0.f;
  #pragma unroll
  for (int i = 0; i < 4; ++i) { float d = v[i] - mu; sq += d*d; }
  red[t] = sq; __syncthreads();
  for (int st = 128; st > 0; st >>= 1) { if (t < st) red[t] += red[t+st]; __syncthreads(); }
  const float rs = rsqrtf(red[0] * (1.0f / DMODEL) + 1e-5f);
  #pragma unroll
  for (int i = 0; i < 4; ++i) {
    int c = t + i*256;
    xnh[(size_t)n*DMODEL + c] = (_Float16)((v[i] - mu) * rs * g[c] + b[c]);
  }
}

// ---------------- f32 -> f16 weight cast ----------------
__global__ void __launch_bounds__(256) cvt_kernel(const float* __restrict__ in,
                                                  _Float16* __restrict__ out, int n) {
  int i = blockIdx.x * 256 + threadIdx.x;
  if (i < n) out[i] = (_Float16)in[i];
}

// ---------------- rotary on q,k,v; emit head-major f16 q/k and transposed f16 v -------
__global__ void __launch_bounds__(64) rotary_kernel(const float* __restrict__ qf,
                                                    const float* __restrict__ kf,
                                                    const float* __restrict__ vf,
                                                    const float* __restrict__ freqs,
                                                    _Float16* __restrict__ qh,
                                                    _Float16* __restrict__ kh,
                                                    _Float16* __restrict__ vt) {
  const int n = blockIdx.x, h = blockIdx.y, dh = threadIdx.x;
  const size_t idx = (size_t)n * DMODEL + h * DHEAD + dh;
  float q = qf[idx], k = kf[idx], v = vf[idx];
  if (dh < ROTD) {
    const float f = freqs[n * ROTD + dh];
    const float c = cosf(f), s = sinf(f);
    // rotate_half: even -> -t[dh+1], odd -> +t[dh-1]
    const size_t pidx = (dh & 1) ? (idx - 1) : (idx + 1);
    const float sgn = (dh & 1) ? 1.f : -1.f;
    q = q * c + sgn * qf[pidx] * s;
    k = k * c + sgn * kf[pidx] * s;
    v = v * c + sgn * vf[pidx] * s;
  }
  const size_t hoff = ((size_t)h * NCTX + n) * DHEAD + dh;
  qh[hoff] = (_Float16)q;
  kh[hoff] = (_Float16)k;
  vt[((size_t)h * DHEAD + dh) * NCTX + n] = (_Float16)v;   // transposed: (h, d, n)
}

// ---------------- generic WMMA GEMM: C = A * B^T (K contiguous on both) --------------
struct GArgs {
  const _Float16* A; unsigned long long sAz; int ldA;
  const _Float16* B0; const _Float16* B1; const _Float16* B2;
  unsigned long long sBz; int ldB;
  float* C0; float* C1; float* C2;
  const float* Af;                   // MODE 2/3: fp32 A, converted to f16 in the loader
  const float* prev; float* dots;    // MODE 1
  const float* bo; const float* xres;// MODE 3
  int K;
};

// MODE 0: QKV projection (z selects Wq/Wk/Wv and qf/kf/vf)
// MODE 1: dots = q k^T * scale + prev, causal mask, write dots (z = head)
// MODE 2: out_heads = attn(f32->f16) * v^T   (z = head, column offset z*64, K truncated at diagonal)
// MODE 3: final = out_heads(f32->f16) * Wo^T + bo + residual
template<int MODE>
__global__ void __launch_bounds__(256) gemm_kernel(GArgs g) {
  const int tid  = threadIdx.x;
  const int lane = tid & 31;
  const int wave = tid >> 5;
  const int wr = wave & 3, wc = wave >> 2;          // 4x2 wave grid -> 64x32 block tile
  const int z = blockIdx.z;
  const int rowBase = blockIdx.y * 64;
  const int colBase = blockIdx.x * 32;
  const int tr = rowBase + wr * 16;
  const int tc = colBase + wc * 16;
  const int m  = lane & 15;                          // A row / B col within 16-tile
  const int kg = lane >> 4;                          // K-half select

  if constexpr (MODE == 1) {
    // block fully above the diagonal: all masked, skip the GEMM entirely
    if (colBase >= rowBase + 64) {
      float* dp = g.dots + (unsigned long long)z * NCTX * NCTX;
      #pragma unroll
      for (int r = 0; r < 8; ++r)
        dp[(unsigned long long)(tr + r + 8*kg) * NCTX + (tc + m)] = NEG_INF;
      return;
    }
  }

  constexpr bool A_F32 = (MODE == 2) || (MODE == 3);
  const _Float16* Ap  = A_F32 ? nullptr : g.A  + g.sAz * (unsigned long long)z;
  const float*    Afp = A_F32 ? g.Af + g.sAz * (unsigned long long)z : nullptr;
  const _Float16* Bp;
  if constexpr (MODE == 0) Bp = (z == 0) ? g.B0 : (z == 1) ? g.B1 : g.B2;
  else                     Bp = g.B0 + g.sBz * (unsigned long long)z;

  int kEnd = g.K;
  if constexpr (MODE == 2) {                         // attn row i is zero past j > i
    int ke = rowBase + 64; kEnd = (ke < g.K) ? ke : g.K;
  }

  const int ldA = g.ldA, ldB = g.ldB;
  v8f acc = {0.f, 0.f, 0.f, 0.f, 0.f, 0.f, 0.f, 0.f};

  for (int k0 = 0; k0 < kEnd; k0 += 64) {
    // ---- A fragments (16x32 f16 layout: lanes 0-15 K 0-7 & 16-23; lanes 16-31 K 8-15 & 24-31)
    v16h a0, a1;
    if constexpr (A_F32) {
      const float* p = Afp + (size_t)(tr + m) * ldA + k0 + 8 * kg;
      float4 u0 = *(const float4*)(p),      u1 = *(const float4*)(p + 4);
      float4 u2 = *(const float4*)(p + 16), u3 = *(const float4*)(p + 20);
      float4 u4 = *(const float4*)(p + 32), u5 = *(const float4*)(p + 36);
      float4 u6 = *(const float4*)(p + 48), u7 = *(const float4*)(p + 52);
      a0 = pack16(u0, u1, u2, u3);
      a1 = pack16(u4, u5, u6, u7);
      __builtin_prefetch(p + 64, 0, 1);
    } else {
      const _Float16* p = Ap + (size_t)(tr + m) * ldA + k0 + 8 * kg;
      AF fa0, fa1;
      fa0.q[0] = *(const uint4*)(p);      fa0.q[1] = *(const uint4*)(p + 16);
      fa1.q[0] = *(const uint4*)(p + 32); fa1.q[1] = *(const uint4*)(p + 48);
      a0 = fa0.v; a1 = fa1.v;
    }
    // ---- B fragments (32x16: lane = col, lanes 0-15 K 0-7 & 16-23; lanes 16-31 K 8-15 & 24-31)
    const _Float16* pb = Bp + (size_t)(tc + m) * ldB + k0 + 8 * kg;
    AF fb0, fb1;
    fb0.q[0] = *(const uint4*)(pb);      fb0.q[1] = *(const uint4*)(pb + 16);
    fb1.q[0] = *(const uint4*)(pb + 32); fb1.q[1] = *(const uint4*)(pb + 48);
    __builtin_prefetch(pb + 64, 0, 1);

    acc = __builtin_amdgcn_wmma_f32_16x16x32_f16(false, a0, false, fb0.v,
                                                 (short)0, acc, false, false);
    acc = __builtin_amdgcn_wmma_f32_16x16x32_f16(false, a1, false, fb1.v,
                                                 (short)0, acc, false, false);
  }

  // ---- epilogue: C/D layout: VGPR r -> row r + 8*(lane>=16), col = lane&15
  #pragma unroll
  for (int r = 0; r < 8; ++r) {
    const int row = tr + r + 8 * kg;
    const int col = tc + m;
    const float cv = acc[r];
    if constexpr (MODE == 0) {
      float* C = (z == 0) ? g.C0 : (z == 1) ? g.C1 : g.C2;
      C[(size_t)row * DMODEL + col] = cv;
    } else if constexpr (MODE == 1) {
      const size_t off = (unsigned long long)z * NCTX * NCTX + (size_t)row * NCTX + col;
      g.dots[off] = (col > row) ? NEG_INF : (cv * QK_SCALE + g.prev[off]);
    } else if constexpr (MODE == 2) {
      g.C0[(size_t)row * DMODEL + z * DHEAD + col] = cv;
    } else {
      const size_t off = (size_t)row * DMODEL + col;
      g.C0[off] = cv + g.bo[col] + g.xres[off];
    }
  }
}

// ---------------- per-row max / sum(exp) over dots ----------------
__global__ void __launch_bounds__(256) rowstats_kernel(const float* __restrict__ dots,
                                                       float* __restrict__ stats) {
  __shared__ float red[256];
  const size_t row = blockIdx.x;
  const float* p = dots + row * NCTX;
  const int t = threadIdx.x;
  float v[8]; float mx = NEG_INF;
  #pragma unroll
  for (int i = 0; i < 8; ++i) { v[i] = p[t + i*256]; mx = fmaxf(mx, v[i]); }
  red[t] = mx; __syncthreads();
  for (int s = 128; s > 0; s >>= 1) { if (t < s) red[t] = fmaxf(red[t], red[t+s]); __syncthreads(); }
  mx = red[0]; __syncthreads();
  float sum = 0.f;
  #pragma unroll
  for (int i = 0; i < 8; ++i) sum += expf(v[i] - mx);
  red[t] = sum; __syncthreads();
  for (int s = 128; s > 0; s >>= 1) { if (t < s) red[t] += red[t+s]; __syncthreads(); }
  if (t == 0) { stats[2*row] = mx; stats[2*row + 1] = red[0]; }
}

// ---------------- attn_map = softmax(dots) ----------------
__global__ void __launch_bounds__(256) attnwrite_kernel(const float* __restrict__ dots,
                                                        const float* __restrict__ stats,
                                                        float* __restrict__ attn) {
  const size_t row = blockIdx.x;
  const float mx = stats[2*row];
  const float inv = 1.0f / stats[2*row + 1];
  const float* p = dots + row * NCTX;
  float* o = attn + row * NCTX;
  const int t = threadIdx.x;
  #pragma unroll
  for (int i = 0; i < 8; ++i) {
    const int c = t + i*256;
    o[c] = expf(p[c] - mx) * inv;
  }
}

extern "C" void kernel_launch(void* const* d_in, const int* in_sizes, int n_in,
                              void* d_out, int out_size, void* d_ws, size_t ws_size,
                              hipStream_t stream) {
  const float* x    = (const float*)d_in[0];
  const float* prev = (const float*)d_in[1];
  const float* rotf = (const float*)d_in[2];
  const float* lng  = (const float*)d_in[3];
  const float* lnb  = (const float*)d_in[4];
  const float* Wq   = (const float*)d_in[5];
  const float* Wk   = (const float*)d_in[6];
  const float* Wv   = (const float*)d_in[7];
  const float* Wo   = (const float*)d_in[8];
  const float* bo   = (const float*)d_in[9];

  char* ws = (char*)d_ws;                                   // ~56.25 MB used
  _Float16* xnh = (_Float16*)(ws);                          // 4 MB
  _Float16* wqh = (_Float16*)(ws + (4ull  << 20));          // 2 MB each
  _Float16* wkh = (_Float16*)(ws + (6ull  << 20));
  _Float16* wvh = (_Float16*)(ws + (8ull  << 20));
  _Float16* woh = (_Float16*)(ws + (10ull << 20));
  float*    qf  = (float*)   (ws + (12ull << 20));          // 8 MB each
  float*    kf  = (float*)   (ws + (20ull << 20));
  float*    vf  = (float*)   (ws + (28ull << 20));
  _Float16* qh  = (_Float16*)(ws + (36ull << 20));          // 4 MB each
  _Float16* kh  = (_Float16*)(ws + (40ull << 20));
  _Float16* vt  = (_Float16*)(ws + (44ull << 20));
  float*    oh  = (float*)   (ws + (48ull << 20));          // 8 MB
  float*  stats = (float*)   (ws + (56ull << 20));          // 256 KB

  float* out  = (float*)d_out;                              // (1,2048,1024)
  float* attn = out  + (size_t)NCTX * DMODEL;               // (1,16,2048,2048)
  float* dots = attn + (size_t)NHEAD * NCTX * NCTX;         // (1,16,2048,2048)

  // 1) LayerNorm + cast
  ln_kernel<<<NCTX, 256, 0, stream>>>(x, lng, lnb, xnh);
  // 2) weight casts
  cvt_kernel<<<4096, 256, 0, stream>>>(Wq, wqh, DMODEL * DMODEL);
  cvt_kernel<<<4096, 256, 0, stream>>>(Wk, wkh, DMODEL * DMODEL);
  cvt_kernel<<<4096, 256, 0, stream>>>(Wv, wvh, DMODEL * DMODEL);
  cvt_kernel<<<4096, 256, 0, stream>>>(Wo, woh, DMODEL * DMODEL);

  // 3) QKV projections (one launch, z selects q/k/v)
  GArgs g0 = {};
  g0.A = xnh; g0.sAz = 0; g0.ldA = DMODEL;
  g0.B0 = wqh; g0.B1 = wkh; g0.B2 = wvh; g0.sBz = 0; g0.ldB = DMODEL;
  g0.C0 = qf; g0.C1 = kf; g0.C2 = vf; g0.K = DMODEL;
  gemm_kernel<0><<<dim3(DMODEL/32, NCTX/64, 3), 256, 0, stream>>>(g0);

  // 4) rotary + repack (q,k head-major f16; v transposed per head)
  rotary_kernel<<<dim3(NCTX, NHEAD), DHEAD, 0, stream>>>(qf, kf, vf, rotf, qh, kh, vt);

  // 5) dots = q k^T * scale + prev, causal mask (writes dots output)
  GArgs g1 = {};
  g1.A = qh; g1.sAz = (unsigned long long)NCTX * DHEAD; g1.ldA = DHEAD;
  g1.B0 = kh; g1.sBz = (unsigned long long)NCTX * DHEAD; g1.ldB = DHEAD;
  g1.prev = prev; g1.dots = dots; g1.K = DHEAD;
  gemm_kernel<1><<<dim3(NCTX/32, NCTX/64, NHEAD), 256, 0, stream>>>(g1);

  // 6) softmax stats + attn_map write
  rowstats_kernel<<<NHEAD * NCTX, 256, 0, stream>>>(dots, stats);
  attnwrite_kernel<<<NHEAD * NCTX, 256, 0, stream>>>(dots, stats, attn);

  // 7) out_heads = attn * v  (fp32 attn converted to f16 in the loader; K stops at diagonal)
  GArgs g2 = {};
  g2.Af = attn; g2.sAz = (unsigned long long)NCTX * NCTX; g2.ldA = NCTX;
  g2.B0 = vt; g2.sBz = (unsigned long long)DHEAD * NCTX; g2.ldB = NCTX;
  g2.C0 = oh; g2.K = NCTX;
  gemm_kernel<2><<<dim3(DHEAD/32, NCTX/64, NHEAD), 256, 0, stream>>>(g2);

  // 8) final projection + bias + residual (fp32 out_heads converted in the loader)
  GArgs g3 = {};
  g3.Af = oh; g3.sAz = 0; g3.ldA = DMODEL;
  g3.B0 = woh; g3.sBz = 0; g3.ldB = DMODEL;
  g3.C0 = out; g3.bo = bo; g3.xres = x; g3.K = DMODEL;
  gemm_kernel<3><<<dim3(DMODEL/32, NCTX/64, 1), 256, 0, stream>>>(g3);
}
